// PAMoE_42356967473829
// MI455X (gfx1250) — matrix operand order repaired and checked by reading, not compiled, not run on hardware
//
#include <hip/hip_runtime.h>
#include <hip/hip_bf16.h>
#include <math.h>
#include <stdint.h>

typedef _Float16 h4  __attribute__((ext_vector_type(4)));
typedef _Float16 h8  __attribute__((ext_vector_type(8)));
typedef _Float16 v16h __attribute__((ext_vector_type(16)));
typedef float    v8f  __attribute__((ext_vector_type(8)));
typedef int      i32x4 __attribute__((ext_vector_type(4)));

#define T_TOK 8192
#define DIM   1024
#define NEXP  8
#define FFN   2048
#define KSEL  1024
#define LDT   40          // LDS tile row stride in halves (32 + 8 pad) = 80B, keeps 16B align
#define TILE_H (128 * LDT)

#if defined(__has_builtin)
#if __has_builtin(__builtin_amdgcn_global_load_async_to_lds_b128) && \
    __has_builtin(__builtin_amdgcn_s_wait_asynccnt)
#define HAS_ASYNC_LDS 1
#endif
#endif
#ifndef HAS_ASYNC_LDS
#define HAS_ASYNC_LDS 0
#endif

#if HAS_ASYNC_LDS
// Builtin expects: (int4 AS(1)* src, int4 AS(3)* dst, imm offset, imm cpol).
// C++ forbids pointer casts across address spaces, so round-trip via uintptr_t
// (generic LDS pointers carry the LDS byte offset in their low 32 bits).
typedef i32x4 __attribute__((address_space(1)))* as1_i4p;
typedef i32x4 __attribute__((address_space(3)))* as3_i4p;
#define ASYNC_LDS_B128(g, l)                                                   \
    __builtin_amdgcn_global_load_async_to_lds_b128(                            \
        (as1_i4p)(uintptr_t)(g), (as3_i4p)(uintptr_t)(l), 0, 0)
#endif

// ---------------------------------------------------------------------------
// 0) zero the scatter-add output region
// ---------------------------------------------------------------------------
__global__ __launch_bounds__(256) void pamoe_zero(float4* __restrict__ p) {
    p[(size_t)blockIdx.x * 256 + threadIdx.x] = make_float4(0.f, 0.f, 0.f, 0.f);
}

// ---------------------------------------------------------------------------
// 0b) fp32 -> f16 weight conversion (one float4 -> h4 per thread)
// ---------------------------------------------------------------------------
__global__ __launch_bounds__(256) void pamoe_cvt(const float4* __restrict__ src,
                                                 h4* __restrict__ dst) {
    const size_t i = (size_t)blockIdx.x * 256 + threadIdx.x;
    const float4 v = src[i];
    h4 o;
    o[0] = (_Float16)v.x; o[1] = (_Float16)v.y; o[2] = (_Float16)v.z; o[3] = (_Float16)v.w;
    dst[i] = o;
}

// ---------------------------------------------------------------------------
// 1) router: logits = x @ rw^T (fp32), softmax probs (fp32). wave-per-token.
// ---------------------------------------------------------------------------
__global__ __launch_bounds__(256) void pamoe_router(
    const float* __restrict__ x, const float* __restrict__ rw,
    float* __restrict__ logits, float* __restrict__ probs) {
    __shared__ float wl[NEXP * DIM];
    const int tid = threadIdx.x;
    for (int i = tid; i < NEXP * DIM; i += 256) wl[i] = rw[i];
    __syncthreads();
    const int wave = tid >> 5, lane = tid & 31;
    const int t = blockIdx.x * 8 + wave;
    const float* xr = x + (size_t)t * DIM;
    float acc[NEXP];
#pragma unroll
    for (int e = 0; e < NEXP; ++e) acc[e] = 0.f;
    for (int d = lane; d < DIM; d += 32) {
        const float xv = xr[d];
#pragma unroll
        for (int e = 0; e < NEXP; ++e) acc[e] += xv * wl[e * DIM + d];
    }
#pragma unroll
    for (int e = 0; e < NEXP; ++e)
#pragma unroll
        for (int off = 16; off > 0; off >>= 1)
            acc[e] += __shfl_xor(acc[e], off, 32);
    if (lane == 0) {
        float mx = acc[0];
#pragma unroll
        for (int e = 1; e < NEXP; ++e) mx = fmaxf(mx, acc[e]);
        float ex[NEXP], s = 0.f;
#pragma unroll
        for (int e = 0; e < NEXP; ++e) { ex[e] = expf(acc[e] - mx); s += ex[e]; }
        const float inv = 1.f / s;
#pragma unroll
        for (int e = 0; e < NEXP; ++e) {
            logits[(size_t)t * NEXP + e] = acc[e];
            probs [(size_t)t * NEXP + e] = ex[e] * inv;
        }
    }
}

// ---------------------------------------------------------------------------
// 2) expert-choice top-K: bitonic sort of 8192 (prob, idx) pairs in 64KB LDS.
// ---------------------------------------------------------------------------
__global__ __launch_bounds__(1024) void pamoe_topk(
    const float* __restrict__ probs, int* __restrict__ sel, float* __restrict__ gate) {
    __shared__ float sv[T_TOK];
    __shared__ int   si[T_TOK];
    const int e = blockIdx.x, tid = threadIdx.x;
    for (int t = tid; t < T_TOK; t += 1024) { sv[t] = probs[(size_t)t * NEXP + e]; si[t] = t; }
    __syncthreads();
    for (unsigned k = 2; k <= T_TOK; k <<= 1) {
        for (unsigned j = k >> 1; j > 0; j >>= 1) {
            for (unsigned i = tid; i < T_TOK; i += 1024) {
                const unsigned ix = i ^ j;
                if (ix > i) {
                    const float va = sv[i], vb = sv[ix];
                    const int   ia = si[i], ib = si[ix];
                    const bool before = (va > vb) || (va == vb && ia < ib);
                    const bool up = ((i & k) == 0);
                    if (up ? !before : before) {
                        sv[i] = vb; sv[ix] = va; si[i] = ib; si[ix] = ia;
                    }
                }
            }
            __syncthreads();
        }
    }
    if (tid < KSEL) {
        sel [e * KSEL + tid] = si[tid];
        gate[e * KSEL + tid] = sv[tid];
    }
}

// ---------------------------------------------------------------------------
// 3) gather selected token rows, fp32 -> f16, contiguous [E*K, D]
// ---------------------------------------------------------------------------
__global__ __launch_bounds__(256) void pamoe_gather(
    const float* __restrict__ x, const int* __restrict__ sel, _Float16* __restrict__ xg) {
    const int r = blockIdx.x;
    const int tok = sel[r];
    const float4 v = ((const float4*)(x + (size_t)tok * DIM))[threadIdx.x];
    h4 o;
    o[0] = (_Float16)v.x; o[1] = (_Float16)v.y; o[2] = (_Float16)v.z; o[3] = (_Float16)v.w;
    ((h4*)(xg + (size_t)r * DIM))[threadIdx.x] = o;
}

// ---------------------------------------------------------------------------
// 4) fused LayerNorm (stats + apply, in place). Wave-per-row; the 4KB row is
//    held in 32 VGPRs so H is read once and written once.
// ---------------------------------------------------------------------------
__global__ __launch_bounds__(256) void pamoe_ln(
    _Float16* __restrict__ hh, const float* __restrict__ ln_g, const float* __restrict__ ln_b) {
    const int wave = threadIdx.x >> 5, lane = threadIdx.x & 31;
    const int r = blockIdx.x * 8 + wave;         // 0 .. E*K-1
    const int e = r >> 10;                        // r / KSEL
    h8* row8 = (h8*)(hh + (size_t)r * FFN);
    h8 v[8];
    float s = 0.f, s2 = 0.f;
#pragma unroll
    for (int q = 0; q < 8; ++q) {
        v[q] = row8[lane + 32 * q];
#pragma unroll
        for (int i = 0; i < 8; ++i) { const float f = (float)v[q][i]; s += f; s2 += f * f; }
    }
#pragma unroll
    for (int off = 16; off > 0; off >>= 1) {
        s  += __shfl_xor(s,  off, 32);
        s2 += __shfl_xor(s2, off, 32);
    }
    const float m   = s * (1.0f / FFN);
    const float var = s2 * (1.0f / FFN) - m * m;
    const float rs  = rsqrtf(var + 1e-5f);
#pragma unroll
    for (int q = 0; q < 8; ++q) {
        const int fb = (lane + 32 * q) * 8;
        const float4 g0 = *(const float4*)(ln_g + (size_t)e * FFN + fb);
        const float4 g1 = *(const float4*)(ln_g + (size_t)e * FFN + fb + 4);
        const float4 b0 = *(const float4*)(ln_b + (size_t)e * FFN + fb);
        const float4 b1 = *(const float4*)(ln_b + (size_t)e * FFN + fb + 4);
        const float gg[8] = {g0.x, g0.y, g0.z, g0.w, g1.x, g1.y, g1.z, g1.w};
        const float bb[8] = {b0.x, b0.y, b0.z, b0.w, b1.x, b1.y, b1.z, b1.w};
        h8 o;
#pragma unroll
        for (int i = 0; i < 8; ++i)
            o[i] = (_Float16)(((float)v[q][i] - m) * rs * gg[i] + bb[i]);
        row8[lane + 32 * q] = o;
    }
}

// ---------------------------------------------------------------------------
// WMMA fragment loaders (wave32, V_WMMA_F32_16X16X32_F16 layouts)
//   A 16x32:  lane l -> row (l&15); halves 0..7 = K kb..kb+7, 8..15 = kb+16..kb+23, kb=(l>>4)*8
//   B 32x16:  lane l -> col (l&15); halves 0..7 = K kb..kb+7, 8..15 = kb+16..kb+23, kb=(l>>4)*8
// ---------------------------------------------------------------------------
__device__ __forceinline__ v16h frag_a(const _Float16* tile, int row0, int lane) {
    const _Float16* p = tile + (size_t)(row0 + (lane & 15)) * LDT + ((lane >> 4) << 3);
    const h8 lo = *(const h8*)p;
    const h8 hi = *(const h8*)(p + 16);
    v16h r;
#pragma unroll
    for (int i = 0; i < 8; ++i) { r[i] = lo[i]; r[i + 8] = hi[i]; }
    return r;
}
__device__ __forceinline__ v16h frag_b(const _Float16* tile, int row0, int lane) {
    const _Float16* p = tile + (size_t)(row0 + (lane & 15)) * LDT + ((lane >> 4) << 3);
    const h8 lo = *(const h8*)p;
    const h8 hi = *(const h8*)(p + 16);
    v16h r;
#pragma unroll
    for (int i = 0; i < 8; ++i) { r[i] = lo[i]; r[i + 8] = hi[i]; }
    return r;
}

// shared epilogue: STAGE 1 -> bias + exact GELU -> f16 H; STAGE 2 -> gated atomic scatter
template <int STAGE>
__device__ __forceinline__ void gemm_epilogue(
    v8f (&acc)[2][4], int e, int m0, int n0, int moff, int noff, int lane,
    const float* bias, const float* gate, const int* sel,
    _Float16* Hout, float* Out) {
    constexpr int Nn = (STAGE == 1) ? FFN : DIM;
    const int mb = (lane >> 4) << 3;
    const int nb = lane & 15;
#pragma unroll
    for (int mi = 0; mi < 2; ++mi) {
#pragma unroll
        for (int v = 0; v < 8; ++v) {
            const int m = m0 + moff + mi * 16 + mb + v;
            float gm = 0.f; int tok = 0;
            if (STAGE == 2) { gm = gate[e * KSEL + m]; tok = sel[e * KSEL + m]; }
#pragma unroll
            for (int ni = 0; ni < 4; ++ni) {
                const int n = n0 + noff + ni * 16 + nb;
                const float xv = acc[mi][ni][v] + bias[e * Nn + n];
                if (STAGE == 1) {
                    const float gl = 0.5f * xv * (1.0f + erff(xv * 0.70710678118654752440f));
                    Hout[((size_t)e * KSEL + m) * FFN + n] = (_Float16)gl;
                } else {
                    atomicAdd(Out + (size_t)tok * DIM + n, xv * gm);
                }
            }
        }
    }
}

// ---------------------------------------------------------------------------
// GEMM (async path): f16 A and f16 W, double-buffered LDS, one chunk ahead via
// GLOBAL_LOAD_ASYNC_TO_LDS_B128 + s_wait_asynccnt (sync copy fallback if the
// builtins are unavailable). 8 waves, 128x128 block tile, 32x64 per wave.
// ---------------------------------------------------------------------------
template <int STAGE>
__global__ __launch_bounds__(256) void pamoe_gemm_async(
    const _Float16* __restrict__ A16, const _Float16* __restrict__ Wh,
    const float* __restrict__ bias,
    const float* __restrict__ gate, const int* __restrict__ sel,
    _Float16* __restrict__ Hout, float* __restrict__ Out) {
    constexpr int Kd = (STAGE == 1) ? DIM : FFN;
    constexpr int NCHUNK = Kd / 32;
    constexpr int Nn = (STAGE == 1) ? FFN : DIM;
    __shared__ _Float16 As[2][TILE_H];
    __shared__ _Float16 Bs[2][TILE_H];

    const int tid = threadIdx.x;
    const int lane = tid & 31, wave = tid >> 5;
    const int e  = blockIdx.z;
    const int m0 = blockIdx.y * 128, n0 = blockIdx.x * 128;
    const int moff = (wave & 3) * 32;
    const int noff = (wave >> 2) * 64;

    const int lrow = tid >> 1;
    const int lcol = (tid & 1) * 16;
    const _Float16* Ag = A16 + ((size_t)e * KSEL + m0 + lrow) * Kd + lcol;
    const _Float16* Bg = Wh  + ((size_t)e * Nn   + n0 + lrow) * Kd + lcol;
    _Float16* Al = &As[0][lrow * LDT + lcol];
    _Float16* Bl = &Bs[0][lrow * LDT + lcol];

    auto issue = [&](int chunk, int buf) {
        const _Float16* ga = Ag + chunk * 32;
        const _Float16* gb = Bg + chunk * 32;
        _Float16* la = Al + buf * TILE_H;
        _Float16* lb = Bl + buf * TILE_H;
#if HAS_ASYNC_LDS
        ASYNC_LDS_B128(ga,     la);
        ASYNC_LDS_B128(ga + 8, la + 8);
        ASYNC_LDS_B128(gb,     lb);
        ASYNC_LDS_B128(gb + 8, lb + 8);
#else
        *(h8*)la       = *(const h8*)ga;
        *(h8*)(la + 8) = *(const h8*)(ga + 8);
        *(h8*)lb       = *(const h8*)gb;
        *(h8*)(lb + 8) = *(const h8*)(gb + 8);
#endif
    };

    v8f acc[2][4];
#pragma unroll
    for (int mi = 0; mi < 2; ++mi)
#pragma unroll
        for (int ni = 0; ni < 4; ++ni)
#pragma unroll
            for (int i = 0; i < 8; ++i) acc[mi][ni][i] = 0.f;

    issue(0, 0);
    for (int i = 0; i < NCHUNK; ++i) {
        const int buf = i & 1;
        if (i + 1 < NCHUNK) issue(i + 1, buf ^ 1);
#if HAS_ASYNC_LDS
        if (i + 1 < NCHUNK) __builtin_amdgcn_s_wait_asynccnt(4);  // chunk i's 4 ops done
        else                __builtin_amdgcn_s_wait_asynccnt(0);
#endif
        __syncthreads();

        v16h af[2], bf[4];
#pragma unroll
        for (int mi = 0; mi < 2; ++mi) af[mi] = frag_a(As[buf], moff + mi * 16, lane);
#pragma unroll
        for (int ni = 0; ni < 4; ++ni) bf[ni] = frag_b(Bs[buf], noff + ni * 16, lane);
#pragma unroll
        for (int mi = 0; mi < 2; ++mi)
#pragma unroll
            for (int ni = 0; ni < 4; ++ni)
                acc[mi][ni] = __builtin_amdgcn_wmma_f32_16x16x32_f16(
                    false, af[mi], false, bf[ni], (short)0, acc[mi][ni], false, false);
        __syncthreads();
    }

    gemm_epilogue<STAGE>(acc, e, m0, n0, moff, noff, lane, bias, gate, sel, Hout, Out);
}

// ---------------------------------------------------------------------------
// GEMM (small-workspace fallback): fp32 weights converted to f16 on the fly.
// ---------------------------------------------------------------------------
template <int STAGE>
__global__ __launch_bounds__(256) void pamoe_gemm_f32w(
    const _Float16* __restrict__ A16, const float* __restrict__ W,
    const float* __restrict__ bias,
    const float* __restrict__ gate, const int* __restrict__ sel,
    _Float16* __restrict__ Hout, float* __restrict__ Out) {
    constexpr int Kd = (STAGE == 1) ? DIM : FFN;
    constexpr int Nn = (STAGE == 1) ? FFN : DIM;
    __shared__ _Float16 As[TILE_H];
    __shared__ _Float16 Bs[TILE_H];

    const int tid = threadIdx.x;
    const int lane = tid & 31, wave = tid >> 5;
    const int e  = blockIdx.z;
    const int m0 = blockIdx.y * 128, n0 = blockIdx.x * 128;
    const int moff = (wave & 3) * 32;
    const int noff = (wave >> 2) * 64;

    const int lrow = tid >> 1;
    const int lcol = (tid & 1) * 16;
    const float*    Wrow = W   + ((size_t)e * Nn   + n0 + lrow) * Kd + lcol;
    const _Float16* Arow = A16 + ((size_t)e * KSEL + m0 + lrow) * Kd + lcol;

    v8f acc[2][4];
#pragma unroll
    for (int mi = 0; mi < 2; ++mi)
#pragma unroll
        for (int ni = 0; ni < 4; ++ni)
#pragma unroll
            for (int i = 0; i < 8; ++i) acc[mi][ni][i] = 0.f;

    for (int k0 = 0; k0 < Kd; k0 += 32) {
        *(h8*)(&As[lrow * LDT + lcol])     = *(const h8*)(Arow + k0);
        *(h8*)(&As[lrow * LDT + lcol + 8]) = *(const h8*)(Arow + k0 + 8);
        {
            const float4* ws4 = (const float4*)(Wrow + k0);
            float tf[16];
#pragma unroll
            for (int q = 0; q < 4; ++q) {
                const float4 v = ws4[q];
                tf[4 * q] = v.x; tf[4 * q + 1] = v.y; tf[4 * q + 2] = v.z; tf[4 * q + 3] = v.w;
            }
            h8 b0, b1;
#pragma unroll
            for (int i = 0; i < 8; ++i) { b0[i] = (_Float16)tf[i]; b1[i] = (_Float16)tf[i + 8]; }
            *(h8*)(&Bs[lrow * LDT + lcol])     = b0;
            *(h8*)(&Bs[lrow * LDT + lcol + 8]) = b1;
        }
        __syncthreads();

        v16h af[2], bf[4];
#pragma unroll
        for (int mi = 0; mi < 2; ++mi) af[mi] = frag_a(As, moff + mi * 16, lane);
#pragma unroll
        for (int ni = 0; ni < 4; ++ni) bf[ni] = frag_b(Bs, noff + ni * 16, lane);
#pragma unroll
        for (int mi = 0; mi < 2; ++mi)
#pragma unroll
            for (int ni = 0; ni < 4; ++ni)
                acc[mi][ni] = __builtin_amdgcn_wmma_f32_16x16x32_f16(
                    false, af[mi], false, bf[ni], (short)0, acc[mi][ni], false, false);
        __syncthreads();
    }

    gemm_epilogue<STAGE>(acc, e, m0, n0, moff, noff, lane, bias, gate, sel, Hout, Out);
}

// ---------------------------------------------------------------------------
extern "C" void kernel_launch(void* const* d_in, const int* in_sizes, int n_in,
                              void* d_out, int out_size, void* d_ws, size_t ws_size,
                              hipStream_t stream) {
    const float* x   = (const float*)d_in[0];  // [B,S,D]
    const float* rw  = (const float*)d_in[1];  // [E,D]
    const float* w1  = (const float*)d_in[2];  // [E,F,D]
    const float* b1  = (const float*)d_in[3];  // [E,F]
    const float* lng = (const float*)d_in[4];  // [E,F]
    const float* lnb = (const float*)d_in[5];  // [E,F]
    const float* w2  = (const float*)d_in[6];  // [E,D,F]
    const float* b2  = (const float*)d_in[7];  // [E,D]

    float* out_res = (float*)d_out;                  // [T, D]
    float* out_log = out_res + (size_t)T_TOK * DIM;  // [T, E]

    char* wsb = (char*)d_ws;
    float* probs = (float*)wsb;            wsb += (size_t)T_TOK * NEXP * 4;
    int*   sel   = (int*)wsb;              wsb += (size_t)NEXP * KSEL * 4;
    float* gate  = (float*)wsb;            wsb += (size_t)NEXP * KSEL * 4;
    _Float16* xg = (_Float16*)wsb;         wsb += (size_t)NEXP * KSEL * DIM * 2;
    _Float16* hh = (_Float16*)wsb;         wsb += (size_t)NEXP * KSEL * FFN * 2;
    _Float16* w1h = (_Float16*)wsb;        wsb += (size_t)NEXP * FFN * DIM * 2;
    _Float16* w2h = (_Float16*)wsb;
    const size_t need_async = ((size_t)T_TOK * NEXP + 2 * NEXP * KSEL) * 4 +
                              ((size_t)NEXP * KSEL * DIM + (size_t)NEXP * KSEL * FFN +
                               2 * (size_t)NEXP * FFN * DIM) * 2;   // ~112.3 MB
    const bool big_ws = (ws_size >= need_async);

    pamoe_zero  <<<(T_TOK * DIM) / (256 * 4), 256, 0, stream>>>((float4*)out_res);
    pamoe_router<<<T_TOK / 8, 256, 0, stream>>>(x, rw, out_log, probs);
    pamoe_topk  <<<NEXP, 1024, 0, stream>>>(probs, sel, gate);
    pamoe_gather<<<NEXP * KSEL, 256, 0, stream>>>(x, sel, xg);

    const dim3 g1(FFN / 128, KSEL / 128, NEXP);
    const dim3 g2(DIM / 128, KSEL / 128, NEXP);
    if (big_ws) {
        pamoe_cvt<<<(NEXP * FFN * DIM) / (256 * 4), 256, 0, stream>>>((const float4*)w1, (h4*)w1h);
        pamoe_cvt<<<(NEXP * FFN * DIM) / (256 * 4), 256, 0, stream>>>((const float4*)w2, (h4*)w2h);
        pamoe_gemm_async<1><<<g1, 256, 0, stream>>>(xg, w1h, b1, gate, sel, hh, out_res);
        pamoe_ln<<<NEXP * KSEL / 8, 256, 0, stream>>>(hh, lng, lnb);
        pamoe_gemm_async<2><<<g2, 256, 0, stream>>>(hh, w2h, b2, gate, sel, hh, out_res);
    } else {
        pamoe_gemm_f32w<1><<<g1, 256, 0, stream>>>(xg, w1, b1, gate, sel, hh, out_res);
        pamoe_ln<<<NEXP * KSEL / 8, 256, 0, stream>>>(hh, lng, lnb);
        pamoe_gemm_f32w<2><<<g2, 256, 0, stream>>>(hh, w2, b2, gate, sel, hh, out_res);
    }
}
